// ProportionalNeuron_78735340471026
// MI455X (gfx1250) — matrix operand order repaired and checked by reading, not compiled, not run on hardware
//
#include <hip/hip_runtime.h>
#include <stdint.h>

// ---------------------------------------------------------------------------
// LIF spiking-neuron scan:
//   Kernel 0 (once):  W fp32 -> bf16 (512 KB, L2-resident afterwards)
//   Per T-chunk:
//     Phase 1: ff = X @ W^T batched over (t,b) rows.
//              bf16 WMMA (v_wmma_f32_16x16x32_bf16), f32 accumulate.
//              Tiles staged with GLOBAL_LOAD_ASYNC_TO_LDS_B128, double
//              buffered, synced with s_wait_asynccnt (ASYNCcnt prefetch).
//     Phase 2: per-(b,f) serial recurrence (memory-bound stream).
// ---------------------------------------------------------------------------

#define Bdim   64
#define Fdim   512
#define Kdim   512
#define TCHUNK 128

typedef __attribute__((ext_vector_type(16))) __bf16 v16bf;
typedef __attribute__((ext_vector_type(8)))  float  v8f;

union FragBF { v16bf v; __bf16 h[16]; uint4 q[2]; };
union Pack8  { __bf16 h[8]; uint4 q; };

// Async memory -> LDS copy, 16 bytes per lane, tracked by ASYNCcnt.
// Generic LDS pointers truncate to the 32-bit LDS offset (ISA 10.2), which is
// exactly the VDST operand of global_load_async_to_lds.
__device__ __forceinline__ void async_copy_b128(const void* lds_dst,
                                                const void* gsrc) {
    uint32_t l = (uint32_t)(uintptr_t)lds_dst;
    asm volatile("global_load_async_to_lds_b128 %0, %1, off"
                 :: "v"(l), "v"(gsrc) : "memory");
}
#define WAIT_ASYNCCNT(n) asm volatile("s_wait_asynccnt %0" :: "n"(n) : "memory")

// ---------------------------------------------------------------------------
// Kernel 0: W fp32 -> bf16 (one shot; removes all W conversion from hot loop)
// ---------------------------------------------------------------------------
__global__ __launch_bounds__(256)
void wcvt_kernel(const float* __restrict__ W, __bf16* __restrict__ Wb) {
    const int i = (blockIdx.x * 256 + threadIdx.x) * 8;
    float4 a = *(const float4*)(W + i);
    float4 b = *(const float4*)(W + i + 4);
    Pack8 p;
    p.h[0] = (__bf16)a.x; p.h[1] = (__bf16)a.y;
    p.h[2] = (__bf16)a.z; p.h[3] = (__bf16)a.w;
    p.h[4] = (__bf16)b.x; p.h[5] = (__bf16)b.y;
    p.h[6] = (__bf16)b.z; p.h[7] = (__bf16)b.w;
    *(uint4*)(Wb + i) = p.q;
}

// ---------------------------------------------------------------------------
// Phase 1: 64x64 output tile per block, 8 waves; wave (mi,nh) = 16(M) x 32(N)
// strip -> two 16x16x32 WMMAs sharing one A fragment per k-step.
// X tile staged as raw fp32 (converted at fragment build), W tile as bf16.
// Grid: (Fdim/64, tCnt)   [rows per chunk = tCnt*B, B=64 -> grid.y = tCnt]
// ---------------------------------------------------------------------------
__global__ __launch_bounds__(256)
void ff_gemm_kernel(const float*  __restrict__ x,   // [T, B, F] fp32
                    const __bf16* __restrict__ Wb,  // [F, K]    bf16
                    float* __restrict__ ff,         // [tCnt*B, F] fp32 (ws)
                    int t0)
{
    __shared__ __align__(16) float  AsF[2][64 * 32]; // X tile fp32, dbl-buf
    __shared__ __align__(16) __bf16 WsH[2][64 * 32]; // W tile bf16, dbl-buf

    const int tid  = threadIdx.x;
    const int lane = tid & 31;
    const int w    = tid >> 5;
    const int mi   = w & 3;                  // M 16-tile within 64
    const int nh   = w >> 2;                 // N 32-half within 64

    const int  fBase   = blockIdx.x * 64;
    const int  rowBase = blockIdx.y * 64;
    const long gRow0   = (long)t0 * Bdim + rowBase;

    // staging: A = 64x32 fp32 (8KB -> 2 async b128/thread),
    //          W = 64x32 bf16 (4KB -> 1 async b128/thread)
    const int ar = tid >> 3, ac = (tid & 7) * 4;
    const int wr = tid >> 2, wc = (tid & 3) * 8;
    const float*  aSrc = x  + (gRow0 + ar) * (long)Fdim + ac;
    const __bf16* wSrc = Wb + (size_t)(fBase + wr) * Kdim + wc;

    const int m  = lane & 15;
    const int kh = lane >> 4;

    v8f acc0 = {}, acc1 = {};

    // prologue: prefetch k0 = 0 into buffer 0 (3 async ops / wave)
    async_copy_b128(&AsF[0][ar * 32 + ac],        aSrc);
    async_copy_b128(&AsF[0][(ar + 32) * 32 + ac], aSrc + 32 * (long)Fdim);
    async_copy_b128(&WsH[0][wr * 32 + wc],        wSrc);

#pragma unroll 2
    for (int it = 0; it < Kdim / 32; ++it) {
        const int buf = it & 1;
        __syncthreads();                       // prev iter's LDS reads done
        if (it + 1 < Kdim / 32) {
            const int k1 = (it + 1) * 32;      // prefetch next k-slice
            async_copy_b128(&AsF[buf ^ 1][ar * 32 + ac],        aSrc + k1);
            async_copy_b128(&AsF[buf ^ 1][(ar + 32) * 32 + ac],
                            aSrc + 32 * (long)Fdim + k1);
            async_copy_b128(&WsH[buf ^ 1][wr * 32 + wc],        wSrc + k1);
            WAIT_ASYNCCNT(3);                  // wait only for current buffer
        } else {
            WAIT_ASYNCCNT(0);
        }
        __syncthreads();                       // everyone's tile landed

        // A fragment (16x32 bf16, ISA layout), converted fp32->bf16 here:
        //   lane<16 : elems 0..7 = K 0..7,  8..15 = K 16..23
        //   lane>=16: elems 0..7 = K 8..15, 8..15 = K 24..31
        const float* arow = &AsF[buf][(mi * 16 + m) * 32];
        float4 f0 = *(const float4*)&arow[kh * 8];
        float4 f1 = *(const float4*)&arow[kh * 8 + 4];
        float4 f2 = *(const float4*)&arow[16 + kh * 8];
        float4 f3 = *(const float4*)&arow[16 + kh * 8 + 4];
        FragBF a;
        a.h[0]  = (__bf16)f0.x; a.h[1]  = (__bf16)f0.y;
        a.h[2]  = (__bf16)f0.z; a.h[3]  = (__bf16)f0.w;
        a.h[4]  = (__bf16)f1.x; a.h[5]  = (__bf16)f1.y;
        a.h[6]  = (__bf16)f1.z; a.h[7]  = (__bf16)f1.w;
        a.h[8]  = (__bf16)f2.x; a.h[9]  = (__bf16)f2.y;
        a.h[10] = (__bf16)f2.z; a.h[11] = (__bf16)f2.w;
        a.h[12] = (__bf16)f3.x; a.h[13] = (__bf16)f3.y;
        a.h[14] = (__bf16)f3.z; a.h[15] = (__bf16)f3.w;

        // B fragments (32x16 bf16): lane<16 K 0..7,16..23, lane>=16 K 8..15,24..31,
        // column N = lane%16 -> k of W row (already bf16), same order as A.
        FragBF b0, b1;
        const __bf16* w0 = &WsH[buf][(nh * 32 + m) * 32 + kh * 8];
        b0.q[0] = *(const uint4*)(w0);
        b0.q[1] = *(const uint4*)(w0 + 16);
        const __bf16* w1 = &WsH[buf][(nh * 32 + 16 + m) * 32 + kh * 8];
        b1.q[0] = *(const uint4*)(w1);
        b1.q[1] = *(const uint4*)(w1 + 16);

        acc0 = __builtin_amdgcn_wmma_f32_16x16x32_bf16(
                   false, a.v, false, b0.v, (short)0, acc0, false, false);
        acc1 = __builtin_amdgcn_wmma_f32_16x16x32_bf16(
                   false, a.v, false, b1.v, (short)0, acc1, false, false);
    }

    // Epilogue: C/D layout -> VGPR j: lanes 0-15 M=j, lanes 16-31 M=j+8
    const int mAdd = (lane < 16) ? 0 : 8;
    const int c0   = fBase + nh * 32 + m;
#pragma unroll
    for (int j = 0; j < 8; ++j) {
        const long row = rowBase + mi * 16 + j + mAdd;
        ff[row * Fdim + c0]      = acc0[j];
        ff[row * Fdim + c0 + 16] = acc1[j];
    }
}

// ---------------------------------------------------------------------------
// Phase 2: per-element serial recurrence over one T-chunk.
// ---------------------------------------------------------------------------
__global__ __launch_bounds__(256)
void lif_scan_kernel(const float* __restrict__ ff,      // [tCnt*B, F]
                     const float* __restrict__ leak_i,
                     const float* __restrict__ leak_v,
                     const float* __restrict__ thresh,
                     float* __restrict__ out,           // [T, B, F]
                     float* __restrict__ state,         // [3, B*F] (ws)
                     int t0, int tCnt)
{
    const int e  = blockIdx.x * blockDim.x + threadIdx.x;
    const int f  = e & (Fdim - 1);
    const int BF = Bdim * Fdim;

    const float li = leak_i[f];
    const float lv = leak_v[f];
    const float th = thresh[f];

    float i, v, z;
    if (t0 == 0) { i = 0.f; v = 0.f; z = 0.f; }
    else { i = state[e]; v = state[BF + e]; z = state[2 * BF + e]; }

    const float* src = ff + e;
    float*       dst = out + (size_t)t0 * BF + e;

#pragma unroll 4
    for (int t = 0; t < tCnt; ++t) {
        const float ffv = src[(size_t)t * BF];
        i = li * i + ffv;
        v = lv * v * (1.f - z) + i;           // reset-to-zero on spike
        z = (v - th) > 0.f ? 1.f : 0.f;       // Heaviside forward
        dst[(size_t)t * BF] = z;
    }

    state[e] = i; state[BF + e] = v; state[2 * BF + e] = z;
}

// ---------------------------------------------------------------------------
extern "C" void kernel_launch(void* const* d_in, const int* in_sizes, int n_in,
                              void* d_out, int out_size, void* d_ws, size_t ws_size,
                              hipStream_t stream)
{
    const float* x  = (const float*)d_in[0];
    const float* W  = (const float*)d_in[1];
    const float* li = (const float*)d_in[2];
    const float* lv = (const float*)d_in[3];
    const float* th = (const float*)d_in[4];
    float* out = (float*)d_out;

    const int T = in_sizes[0] / (Bdim * Fdim);

    // ws layout: ff [TCHUNK*B*F f32] | state [3*B*F f32] | Wb [F*K bf16]
    float*  ffBuf = (float*)d_ws;
    float*  state = ffBuf + (size_t)TCHUNK * Bdim * Fdim;
    __bf16* Wb    = (__bf16*)(state + (size_t)3 * Bdim * Fdim);

    wcvt_kernel<<<dim3((Fdim * Kdim) / (256 * 8)), 256, 0, stream>>>(W, Wb);

    for (int t0 = 0; t0 < T; t0 += TCHUNK) {
        const int tCnt = (T - t0 < TCHUNK) ? (T - t0) : TCHUNK;

        dim3 g1(Fdim / 64, tCnt);
        ff_gemm_kernel<<<g1, 256, 0, stream>>>(x, Wb, ffBuf, t0);

        dim3 g2((Bdim * Fdim) / 256);
        lif_scan_kernel<<<g2, 256, 0, stream>>>(ffBuf, li, lv, th, out, state,
                                                t0, tCnt);
    }
}
